// EntityAttentionLayer_7352984011361
// MI455X (gfx1250) — matrix operand order repaired and checked by reading, not compiled, not run on hardware
//
#include <hip/hip_runtime.h>

typedef __attribute__((ext_vector_type(16))) __bf16 v16bf;
typedef __attribute__((ext_vector_type(8)))  __bf16 v8bf;
typedef __attribute__((ext_vector_type(8)))  float  v8f;
typedef __attribute__((ext_vector_type(4)))  float  v4f;

#define BS      512
#define NE      256
#define NQ      64
#define IN_DIM  512
#define EMBED   512
#define OUT_DIM 512
#define NH      8
#define HD      64
#define QKV_N   1536

static __device__ __forceinline__ int lane_id() { return threadIdx.x & 31; }

static __device__ __forceinline__ v8f wmma_bf16(v16bf a, v16bf b, v8f c) {
    // D = A(16x32) * B(32x16) + C(16x16), f32 accumulate
    return __builtin_amdgcn_wmma_f32_16x16x32_bf16(false, a, false, b, (short)0, c, false, false);
}

// ---- Fragment loaders (ISA 7.12.2 layouts, wave32), vectorized ----

// A fragment 16x32 (MxK), row-major bf16 source, row stride ld (multiple of 8).
static __device__ __forceinline__ v16bf load_a_bf16(const __bf16* src, int ld) {
    int lane = lane_id();
    int m  = lane & 15;
    int kb = (lane & 16) ? 8 : 0;
    const __bf16* p = src + (size_t)m * ld + kb;
    v8bf lo = *(const v8bf*)(p);
    v8bf hi = *(const v8bf*)(p + 16);
    v16bf a;
#pragma unroll
    for (int i = 0; i < 8; ++i) { a[i] = lo[i]; a[8 + i] = hi[i]; }
    return a;
}

// A fragment 16x32 (MxK), row-major f32 source -> 4 x b128 + packed cvt.
static __device__ __forceinline__ v16bf load_a_f32(const float* src, int ld) {
    int lane = lane_id();
    int m  = lane & 15;
    int kb = (lane & 16) ? 8 : 0;
    const float* p = src + (size_t)m * ld + kb;
    v4f x0 = *(const v4f*)(p);
    v4f x1 = *(const v4f*)(p + 4);
    v4f x2 = *(const v4f*)(p + 16);
    v4f x3 = *(const v4f*)(p + 20);
    v16bf a;
#pragma unroll
    for (int i = 0; i < 4; ++i) {
        a[i]      = (__bf16)x0[i];
        a[4 + i]  = (__bf16)x1[i];
        a[8 + i]  = (__bf16)x2[i];
        a[12 + i] = (__bf16)x3[i];
    }
    return a;
}

// B fragment 32x16 (KxN) from an N-MAJOR (transposed) bf16 source:
// element (k,n) = src[n*ld + k].  Per lane: k = kb..kb+7 and kb+16..kb+23 (kb = 8*half) -> 2 x b128.
static __device__ __forceinline__ v16bf load_bT_bf16(const __bf16* src, int ld) {
    int lane = lane_id();
    int n  = lane & 15;
    int kb = (lane & 16) ? 8 : 0;
    const __bf16* p = src + (size_t)n * ld + kb;
    v8bf lo = *(const v8bf*)(p);
    v8bf hi = *(const v8bf*)(p + 16);
    v16bf b;
#pragma unroll
    for (int i = 0; i < 8; ++i) { b[i] = lo[i]; b[8 + i] = hi[i]; }
    return b;
}

// ---------------------------------------------------------------------------
// Kernel 0: transpose + convert a f32 weight [rows x cols] into bf16 [cols x rows]
// ---------------------------------------------------------------------------
__global__ __launch_bounds__(256) void cvtT_kernel(const float* __restrict__ src,
                                                   __bf16* __restrict__ dst,
                                                   int rows, int cols) {
    int idx = blockIdx.x * 256 + threadIdx.x;
    if (idx < rows * cols) {
        int r = idx / cols;
        int c = idx - r * cols;
        dst[(size_t)c * rows + r] = (__bf16)src[idx];
    }
}

// ---------------------------------------------------------------------------
// Kernel 1: QKV projection.  Block = 128 thr (4 waves); block tile 64x128;
// wave tile 16x128 (1 A frag x 8 B tiles = 8 WMMAs per k-step).
// Jobs/batch: Q: 1x4 (rows 0..63) | K: 4x4 | V: 4x4  = 36.
// V is stored TRANSPOSED (Vt[b][col][entity]).
// ---------------------------------------------------------------------------
__global__ __launch_bounds__(128) void qkv_kernel(const float* __restrict__ ent,
                                                  const __bf16* __restrict__ WinT,
                                                  __bf16* __restrict__ Qb,
                                                  __bf16* __restrict__ Kb,
                                                  __bf16* __restrict__ Vt) {
    int b = blockIdx.x / 36;
    int j = blockIdx.x % 36;
    int wave = threadIdx.x >> 5;

    int m0, n0, coloff, isV = 0;
    __bf16* dst;
    if (j < 4) {                 // Q: only first NQ entity rows
        m0 = 0; n0 = j * 128; coloff = 0;
        dst = Qb + (size_t)b * NQ * EMBED;
    } else if (j < 20) {         // K
        int jj = j - 4;
        m0 = (jj >> 2) * 64; n0 = (jj & 3) * 128; coloff = EMBED;
        dst = Kb + (size_t)b * NE * EMBED;
    } else {                     // V (stored transposed)
        int jj = j - 20;
        m0 = (jj >> 2) * 64; n0 = (jj & 3) * 128; coloff = 2 * EMBED;
        dst = Vt + (size_t)b * EMBED * NE;
        isV = 1;
    }

    const float*  A  = ent + (size_t)b * NE * IN_DIM + (size_t)(m0 + wave * 16) * IN_DIM;
    const __bf16* Bw = WinT + (size_t)(coloff + n0) * IN_DIM;

    v8f acc[8] = {};
    v16bf a = load_a_f32(A, IN_DIM);
    for (int kk = 0; kk < IN_DIM; kk += 32) {
        v16bf a_cur = a;
        if (kk + 32 < IN_DIM) a = load_a_f32(A + kk + 32, IN_DIM);   // prefetch next A
#pragma unroll
        for (int t = 0; t < 8; ++t) {
            v16bf bf = load_bT_bf16(Bw + (size_t)(t * 16) * IN_DIM + kk, IN_DIM);
            acc[t] = wmma_bf16(a_cur, bf, acc[t]);
        }
    }

    int lane = lane_id();
    int n  = lane & 15;
    int mb = (lane & 16) ? 8 : 0;
    if (!isV) {
#pragma unroll
        for (int t = 0; t < 8; ++t)
#pragma unroll
            for (int r = 0; r < 8; ++r)
                dst[(size_t)(m0 + wave * 16 + mb + r) * EMBED + n0 + t * 16 + n] = (__bf16)acc[t][r];
    } else {
#pragma unroll
        for (int t = 0; t < 8; ++t)
#pragma unroll
            for (int r = 0; r < 8; ++r)
                dst[(size_t)(n0 + t * 16 + n) * NE + (m0 + wave * 16 + mb + r)] = (__bf16)acc[t][r];
    }
}

// ---------------------------------------------------------------------------
// Kernel 2: attention for one (batch, head).  Block = 256 threads (8 waves).
// Phase 1: logits(64x256) = q @ k^T / 8; wave tile 16x128 (8 WMMAs / A frag).
// Phase 2: masked softmax per row (wave32 shuffle reductions), back into LDS.
// Phase 3: attn(64x64) = w @ v; wave tile 16x32 (V transposed B frags).
// ---------------------------------------------------------------------------
__global__ __launch_bounds__(256) void attn_kernel(const __bf16* __restrict__ Qb,
                                                   const __bf16* __restrict__ Kb,
                                                   const __bf16* __restrict__ Vt,
                                                   const int* __restrict__ pre_mask,
                                                   __bf16* __restrict__ Ab) {
    __shared__ float slog[NQ][NE];   // 64*256*4 = 64 KB

    int b = blockIdx.x / NH;
    int h = blockIdx.x % NH;
    int wave = threadIdx.x >> 5;
    int lane = lane_id();
    int n  = lane & 15;
    int mb = (lane & 16) ? 8 : 0;

    const __bf16* q  = Qb + (size_t)b * NQ * EMBED + h * HD;       // row stride EMBED
    const __bf16* k  = Kb + (size_t)b * NE * EMBED + h * HD;       // row stride EMBED
    const __bf16* vt = Vt + ((size_t)b * EMBED + h * HD) * NE;     // [col][entity], stride NE

    // ---- Phase 1: wave = (mt, ng): rows [16mt,16mt+16), cols [128ng, 128ng+128)
    {
        int mt = wave >> 1;
        int ng = wave & 1;
        v8f acc[8] = {};
#pragma unroll
        for (int kk = 0; kk < HD; kk += 32) {
            v16bf a = load_a_bf16(q + (size_t)(mt * 16) * EMBED + kk, EMBED);
#pragma unroll
            for (int t = 0; t < 8; ++t) {
                v16bf bb = load_bT_bf16(k + (size_t)(ng * 128 + t * 16) * EMBED + kk, EMBED);
                acc[t] = wmma_bf16(a, bb, acc[t]);
            }
        }
#pragma unroll
        for (int t = 0; t < 8; ++t)
#pragma unroll
            for (int r = 0; r < 8; ++r)
                slog[mt * 16 + mb + r][ng * 128 + t * 16 + n] = acc[t][r] * 0.125f;  // 1/sqrt(64)
    }
    __syncthreads();

    // ---- Phase 2: masked softmax, one row per wave iteration (8 elems/lane)
    const int* pm = pre_mask + (size_t)b * NQ * NE;
    for (int row = wave; row < NQ; row += 8) {
        float x[8];
        float mx = -__builtin_inff();
#pragma unroll
        for (int i = 0; i < 8; ++i) {
            int e = lane + 32 * i;
            float val = slog[row][e];
            if (pm[row * NE + e] != 0) val = -__builtin_inff();  // mask==1 -> exclude
            x[i] = val;
            mx = fmaxf(mx, val);
        }
#pragma unroll
        for (int off = 16; off > 0; off >>= 1)
            mx = fmaxf(mx, __shfl_xor(mx, off, 32));

        float sum = 0.0f;
        if (mx == -__builtin_inff()) {
            // fully-masked row: softmax -> NaN -> 0 per reference
#pragma unroll
            for (int i = 0; i < 8; ++i) x[i] = 0.0f;
            sum = 1.0f;
        } else {
#pragma unroll
            for (int i = 0; i < 8; ++i) {
                x[i] = __expf(x[i] - mx);   // exp(-inf) = 0 for masked entries
                sum += x[i];
            }
#pragma unroll
            for (int off = 16; off > 0; off >>= 1)
                sum += __shfl_xor(sum, off, 32);
        }
        float inv = 1.0f / sum;
#pragma unroll
        for (int i = 0; i < 8; ++i)
            slog[row][lane + 32 * i] = x[i] * inv;
    }
    __syncthreads();

    // ---- Phase 3: attn = w(64x256) @ v(256x64); wave = (mt, nh): 16x32 tile
    {
        int mt = wave >> 1;
        int nh2 = wave & 1;
        v8f acc[2] = {};
        for (int kk = 0; kk < NE; kk += 32) {
            v16bf a = load_a_f32(&slog[mt * 16][0] + kk, NE);
#pragma unroll
            for (int t = 0; t < 2; ++t) {
                v16bf bb = load_bT_bf16(vt + (size_t)((nh2 * 2 + t) * 16) * NE + kk, NE);
                acc[t] = wmma_bf16(a, bb, acc[t]);
            }
        }
        __bf16* dst = Ab + (size_t)b * NQ * EMBED + h * HD;
#pragma unroll
        for (int t = 0; t < 2; ++t)
#pragma unroll
            for (int r = 0; r < 8; ++r)
                dst[(size_t)(mt * 16 + mb + r) * EMBED + (nh2 * 2 + t) * 16 + n] = (__bf16)acc[t][r];
    }
}

// ---------------------------------------------------------------------------
// Kernel 3: out = attn @ W_out + b_out, post-masked.  Block tile 64x128,
// wave tile 16x128; grid = (32768/64) * (512/128) = 2048 blocks.
// ---------------------------------------------------------------------------
__global__ __launch_bounds__(128) void out_kernel(const __bf16* __restrict__ Ab,
                                                  const __bf16* __restrict__ WoutT,
                                                  const float* __restrict__ bout,
                                                  const int* __restrict__ post_mask,
                                                  float* __restrict__ out) {
    int mt = blockIdx.x >> 2;      // 0..511, 64-row tile over 32768 rows
    int nt = blockIdx.x & 3;       // 0..3
    int m0 = mt * 64;
    int n0 = nt * 128;
    int wave = threadIdx.x >> 5;

    const __bf16* A  = Ab + (size_t)(m0 + wave * 16) * EMBED;
    const __bf16* Bw = WoutT + (size_t)n0 * EMBED;

    v8f acc[8] = {};
    v16bf a = load_a_bf16(A, EMBED);
    for (int kk = 0; kk < EMBED; kk += 32) {
        v16bf a_cur = a;
        if (kk + 32 < EMBED) a = load_a_bf16(A + kk + 32, EMBED);   // prefetch next A
#pragma unroll
        for (int t = 0; t < 8; ++t) {
            v16bf bf = load_bT_bf16(Bw + (size_t)(t * 16) * EMBED + kk, EMBED);
            acc[t] = wmma_bf16(a_cur, bf, acc[t]);
        }
    }

    int lane = lane_id();
    int n  = lane & 15;
    int mb = (lane & 16) ? 8 : 0;
#pragma unroll
    for (int t = 0; t < 8; ++t) {
#pragma unroll
        for (int r = 0; r < 8; ++r) {
            int gr = m0 + wave * 16 + mb + r;           // global row = b*NQ + qi
            int gc = n0 + t * 16 + n;
            int bb = gr / NQ;
            int qi = gr - bb * NQ;
            float val = acc[t][r] + bout[gc];
            if (post_mask[bb * NQ + qi] != 0) val = 0.0f;
            out[(size_t)gr * OUT_DIM + gc] = val;
        }
    }
}

// ---------------------------------------------------------------------------
extern "C" void kernel_launch(void* const* d_in, const int* in_sizes, int n_in,
                              void* d_out, int out_size, void* d_ws, size_t ws_size,
                              hipStream_t stream) {
    const float* ent       = (const float*)d_in[0];
    const int*   pre_mask  = (const int*)d_in[1];
    const int*   post_mask = (const int*)d_in[2];
    const float* Win       = (const float*)d_in[3];
    const float* Wout      = (const float*)d_in[4];
    const float* bout      = (const float*)d_in[5];
    float*       out       = (float*)d_out;

    // Workspace (bf16): Q 32MB | K 128MB | Vt 128MB | attn 32MB | WinT 1.5MB | WoutT 0.5MB
    char* ws = (char*)d_ws;
    size_t qBytes = (size_t)BS * NQ * EMBED * sizeof(__bf16);
    size_t kBytes = (size_t)BS * NE * EMBED * sizeof(__bf16);
    __bf16* Qb    = (__bf16*)(ws);
    __bf16* Kb    = (__bf16*)(ws + qBytes);
    __bf16* Vt    = (__bf16*)(ws + qBytes + kBytes);
    __bf16* Ab    = (__bf16*)(ws + qBytes + 2 * kBytes);
    __bf16* WinT  = (__bf16*)(ws + 2 * qBytes + 2 * kBytes);
    __bf16* WoutT = (__bf16*)(ws + 2 * qBytes + 2 * kBytes + (size_t)IN_DIM * QKV_N * 2);

    cvtT_kernel<<<(IN_DIM * QKV_N + 255) / 256, 256, 0, stream>>>(Win, WinT, IN_DIM, QKV_N);
    cvtT_kernel<<<(EMBED * OUT_DIM + 255) / 256, 256, 0, stream>>>(Wout, WoutT, EMBED, OUT_DIM);

    qkv_kernel<<<BS * 36, 128, 0, stream>>>(ent, WinT, Qb, Kb, Vt);
    attn_kernel<<<BS * NH, 256, 0, stream>>>(Qb, Kb, Vt, pre_mask, Ab);
    out_kernel<<<(BS * NQ / 64) * (OUT_DIM / 128), 128, 0, stream>>>(Ab, WoutT, bout, post_mask, out);
}
